// DynamicFusionModule_41764261987283
// MI455X (gfx1250) — matrix operand order repaired and checked by reading, not compiled, not run on hardware
//
#include <hip/hip_runtime.h>
#include <hip/hip_bf16.h>
#include <math.h>

// ---------------------------------------------------------------------------
// Problem constants (match reference)
// ---------------------------------------------------------------------------
#define DIM        384
#define HEADS      4
#define HDIM       96
#define NTOK       1024          // 32*32
#define BATCH      8
#define ROWS       (BATCH*NTOK)  // 8192 tokens
#define AGENT_IN   768
#define AGENT_HID  512
#define QKV_DIM    1152
#define FFN_HID    1536
#define MIN_TOK    64
#define NEG_INF_F  (-1000000000.0f)
#define F32_NINF   (-3.402823466e38f)

typedef __attribute__((ext_vector_type(16))) __bf16 v16bf;
typedef __attribute__((ext_vector_type(2)))  __bf16 v2bf;
typedef __attribute__((ext_vector_type(8)))  float  v8f;
typedef __attribute__((ext_vector_type(4)))  unsigned int v4u;

// Epilogue modes for the generic WMMA GEMM
#define EPI_BIAS  0   // C = A@W^T + bias (+residual)
#define EPI_GELU  1   // C = gelu(A@W^T + bias)
#define EPI_SILU  2   // C = silu(A@W^T + bias)
#define EPI_MASK  3   // C = valid[col] ? scale*(A@W^T) : NEG_INF   (bias ptr = int* valid)
#define EPI_PLAIN 4   // C = A@W^T (+residual)

// pack two fp32 into one dword of bf16 -> single v_cvt_pk_bf16_f32
static __device__ __forceinline__ unsigned pack2(float lo, float hi)
{
    v2bf v;
    v[0] = (__bf16)lo;
    v[1] = (__bf16)hi;
    return __builtin_bit_cast(unsigned, v);
}

struct B32x2 { v4u lo, hi; };
// 32-byte LDS fragment load as 2x ds_load_b128 (16B-aligned rows)
static __device__ __forceinline__ v16bf load_frag(const __bf16* p)
{
    B32x2 t;
    t.lo = *(const v4u*)p;
    t.hi = *(const v4u*)(p + 8);
    return __builtin_bit_cast(v16bf, t);
}

// ---------------------------------------------------------------------------
// Generic bf16 WMMA GEMM:  C[M,N] = act(A[M,K] @ B[N,K]^T + bias) + residual
//   A element (m,k) at A + m*lda + k          (k contiguous)
//   B element (n,k) at B + n*ldbn + k*ldbk    (KCONTIG => ldbk==1)
//   C element (m,n) at C + m*ldc + n
// 256 threads = 8 wave32; block tile 128x128; wave tile 32x64.
// K in steps of 32 via v_wmma_f32_16x16x32_bf16. Software-pipelined staging.
// LDS rows hold bf16 with stride 40 (80B): 16B aligned, bank-conflict free.
// A rows are K-permuted [0..7,16..23,8..15,24..31] so each lane's fragment
// is a contiguous 32B chunk per the ISA 16-bit A-matrix layout.
// ---------------------------------------------------------------------------
template<int EPI, bool KCONTIG>
__global__ __launch_bounds__(256)
void gemm_wmma(const float* __restrict__ A, int lda,
               const float* __restrict__ B, long long ldbn, long long ldbk,
               float* __restrict__ C, int ldc,
               const float* __restrict__ bias,      // EPI_MASK: (const int*) valid mask
               const float* __restrict__ residual,  // may be null; layout of C
               int M, int N, int K, float scale)
{
    __shared__ __bf16 As[128][40];
    __shared__ __bf16 Bs[128][40];

    const int tid  = threadIdx.x;
    const int lane = tid & 31;
    const int wave = tid >> 5;
    const int half = lane >> 4;
    const int l16  = lane & 15;
    const int wm   = (wave >> 1) * 32;   // wave row offset within tile
    const int wn   = (wave & 1) * 64;    // wave col offset within tile

    const int m0 = blockIdx.y * 128;
    const int n0 = blockIdx.x * 128;

    // staging coordinates: thread -> (row, 4 consecutive k)
    const int sr  = tid >> 3;            // 0..31 (+32 per i)
    const int sc4 = (tid & 7) * 4;       // 0,4,...,28
    const int blk = sc4 >> 3;            // k-group of 8
    const int posA = ((((blk & 1) << 1) | (blk >> 1)) << 3) + (sc4 & 7); // permuted

    v8f acc[2][4];
    #pragma unroll
    for (int mi = 0; mi < 2; ++mi)
        #pragma unroll
        for (int nt = 0; nt < 4; ++nt)
            #pragma unroll
            for (int e = 0; e < 8; ++e) acc[mi][nt][e] = 0.0f;

    float4 a4[4], b4[4];

    auto load_tiles = [&](int k0) {
        #pragma unroll
        for (int i = 0; i < 4; ++i) {
            int r  = sr + i * 32;
            int gm = min(m0 + r, M - 1);           // clamp: branch-free
            int gn = min(n0 + r, N - 1);
            a4[i] = *(const float4*)&A[(long long)gm * lda + (k0 + sc4)];
            if (KCONTIG) {
                b4[i] = *(const float4*)&B[(long long)gn * ldbn + (k0 + sc4)];
            } else {
                const float* bp = B + (long long)gn * ldbn + (long long)(k0 + sc4) * ldbk;
                b4[i].x = bp[0];
                b4[i].y = bp[ldbk];
                b4[i].z = bp[2 * ldbk];
                b4[i].w = bp[3 * ldbk];
            }
        }
    };
    auto store_tiles = [&]() {
        #pragma unroll
        for (int i = 0; i < 4; ++i) {
            int r = sr + i * 32;
            uint2 pa; pa.x = pack2(a4[i].x, a4[i].y); pa.y = pack2(a4[i].z, a4[i].w);
            *(uint2*)&As[r][posA] = pa;            // ds_store_b64
            uint2 pb; pb.x = pack2(b4[i].x, b4[i].y); pb.y = pack2(b4[i].z, b4[i].w);
            *(uint2*)&Bs[r][posA] = pb;
        }
    };

    load_tiles(0);
    for (int k0 = 0; k0 < K; k0 += 32) {
        store_tiles();
        __syncthreads();
        if (k0 + 32 < K) load_tiles(k0 + 32);      // overlap HBM with WMMA burst

        v16bf af0 = load_frag(&As[wm +      l16][half * 16]);
        v16bf af1 = load_frag(&As[wm + 16 + l16][half * 16]);
        #pragma unroll
        for (int nt = 0; nt < 4; ++nt) {
            v16bf bf = load_frag(&Bs[wn + nt * 16 + l16][half * 16]);
            acc[0][nt] = __builtin_amdgcn_wmma_f32_16x16x32_bf16(
                false, af0, false, bf, (short)0, acc[0][nt], false, false);
            acc[1][nt] = __builtin_amdgcn_wmma_f32_16x16x32_bf16(
                false, af1, false, bf, (short)0, acc[1][nt], false, false);
        }
        __syncthreads();
    }

    // ---- epilogue: C/D layout VGPR v -> row = +v + 8*half, col = +l16 ----
    #pragma unroll
    for (int mi = 0; mi < 2; ++mi) {
        #pragma unroll
        for (int nt = 0; nt < 4; ++nt) {
            int col = n0 + wn + nt * 16 + l16;
            if (col >= N) continue;
            float bv = 0.0f;
            bool kvalid = true;
            if (EPI == EPI_BIAS || EPI == EPI_GELU || EPI == EPI_SILU) bv = bias[col];
            if (EPI == EPI_MASK) kvalid = ((const int*)bias)[col] != 0;
            #pragma unroll
            for (int v = 0; v < 8; ++v) {
                int row = m0 + wm + mi * 16 + half * 8 + v;
                if (row >= M) continue;
                float x = acc[mi][nt][v];
                if (EPI == EPI_BIAS)  x += bv;
                if (EPI == EPI_GELU) {
                    x += bv;
                    float x3 = x * x * x;
                    x = 0.5f * x * (1.0f + tanhf(0.7978845608028654f * (x + 0.044715f * x3)));
                }
                if (EPI == EPI_SILU) { x += bv; x = x / (1.0f + __expf(-x)); }
                if (EPI == EPI_MASK)  x = kvalid ? x * scale : NEG_INF_F;
                if (residual) x += residual[(long long)row * ldc + col];
                C[(long long)row * ldc + col] = x;
            }
        }
    }
}

// ---------------------------------------------------------------------------
// (B,C,N) -> (B,N,C) flatten for both modalities + agent input concat
// ---------------------------------------------------------------------------
__global__ __launch_bounds__(256)
void prep_kernel(const float* __restrict__ f_ir, const float* __restrict__ f_vis,
                 float* __restrict__ flat_ir, float* __restrict__ flat_vis,
                 float* __restrict__ xagent)
{
    long long i = (long long)blockIdx.x * 256 + threadIdx.x;
    const long long TOT = (long long)BATCH * DIM * NTOK;
    if (i >= TOT) return;
    int n = (int)(i % NTOK);
    long long t = i / NTOK;
    int c = (int)(t % DIM);
    int b = (int)(t / DIM);
    float vi = f_ir[i], vv = f_vis[i];
    long long row = (long long)b * NTOK + n;
    flat_ir[row * DIM + c]  = vi;
    flat_vis[row * DIM + c] = vv;
    xagent[row * AGENT_IN + c]       = vi;
    xagent[row * AGENT_IN + DIM + c] = vv;
}

// ---------------------------------------------------------------------------
// LayerNorm over last dim (384). One 128-thread block per row.
// ---------------------------------------------------------------------------
__global__ __launch_bounds__(128)
void ln_kernel(const float* __restrict__ x, const float* __restrict__ g,
               const float* __restrict__ bb, float* __restrict__ y)
{
    const int tid = threadIdx.x;
    const float* xr = x + (long long)blockIdx.x * DIM;
    float v[3]; float s = 0.0f;
    #pragma unroll
    for (int i = 0; i < 3; ++i) { v[i] = xr[tid + i * 128]; s += v[i]; }
    __shared__ float red[4];
    #pragma unroll
    for (int m = 16; m >= 1; m >>= 1) s += __shfl_xor(s, m, 32);
    if ((tid & 31) == 0) red[tid >> 5] = s;
    __syncthreads();
    float mean = (red[0] + red[1] + red[2] + red[3]) * (1.0f / DIM);
    float q = 0.0f;
    #pragma unroll
    for (int i = 0; i < 3; ++i) { float d = v[i] - mean; q += d * d; }
    #pragma unroll
    for (int m = 16; m >= 1; m >>= 1) q += __shfl_xor(q, m, 32);
    __syncthreads();
    if ((tid & 31) == 0) red[tid >> 5] = q;
    __syncthreads();
    float var = (red[0] + red[1] + red[2] + red[3]) * (1.0f / DIM);
    float rs = rsqrtf(var + 1e-5f);
    #pragma unroll
    for (int i = 0; i < 3; ++i) {
        int c = tid + i * 128;
        y[(long long)blockIdx.x * DIM + c] = (v[i] - mean) * rs * g[c] + bb[c];
    }
}

// ---------------------------------------------------------------------------
// Row softmax over 1024 (in place). One 256-thread block per row.
// ---------------------------------------------------------------------------
__global__ __launch_bounds__(256)
void softmax_kernel(float* __restrict__ S)
{
    float* row = S + (long long)blockIdx.x * NTOK;
    const int tid = threadIdx.x;
    __shared__ float red[8];
    float v[4]; float mx = F32_NINF;
    #pragma unroll
    for (int i = 0; i < 4; ++i) { v[i] = row[tid + i * 256]; mx = fmaxf(mx, v[i]); }
    #pragma unroll
    for (int m = 16; m >= 1; m >>= 1) mx = fmaxf(mx, __shfl_xor(mx, m, 32));
    if ((tid & 31) == 0) red[tid >> 5] = mx;
    __syncthreads();
    mx = red[0];
    #pragma unroll
    for (int i = 1; i < 8; ++i) mx = fmaxf(mx, red[i]);
    float s = 0.0f;
    #pragma unroll
    for (int i = 0; i < 4; ++i) { v[i] = __expf(v[i] - mx); s += v[i]; }
    #pragma unroll
    for (int m = 16; m >= 1; m >>= 1) s += __shfl_xor(s, m, 32);
    __syncthreads();
    if ((tid & 31) == 0) red[tid >> 5] = s;
    __syncthreads();
    s = 0.0f;
    #pragma unroll
    for (int i = 0; i < 8; ++i) s += red[i];
    float inv = 1.0f / s;
    #pragma unroll
    for (int i = 0; i < 4; ++i) row[tid + i * 256] = v[i] * inv;
}

// ---------------------------------------------------------------------------
// logits[m] = dot(H[m,:512], w2) + b2 ; one wave32 per row
// ---------------------------------------------------------------------------
__global__ __launch_bounds__(256)
void logits_kernel(const float* __restrict__ H, const float* __restrict__ w2,
                   const float* __restrict__ b2, float* __restrict__ out)
{
    int row  = blockIdx.x * 8 + (threadIdx.x >> 5);
    int lane = threadIdx.x & 31;
    const float* h = H + (long long)row * AGENT_HID;
    float s = 0.0f;
    for (int i = lane; i < AGENT_HID; i += 32) s += h[i] * w2[i];
    #pragma unroll
    for (int m = 16; m >= 1; m >>= 1) s += __shfl_xor(s, m, 32);
    if (lane == 0) out[row] = s + b2[0];
}

// ---------------------------------------------------------------------------
// valid mask: per batch, cnt = #(lf>0); kth = 64th largest (iterative top-k)
// valid = cnt>=64 ? (lf>0) : (lf>=kth)
// ---------------------------------------------------------------------------
__global__ __launch_bounds__(256)
void valid_kernel(const float* __restrict__ logits, int* __restrict__ valid)
{
    int b = blockIdx.x;
    int tid = threadIdx.x;
    __shared__ float vals[NTOK];
    __shared__ float rv[256];
    __shared__ int   ri[256];
    __shared__ int   scnt;
    __shared__ float kth_s;
    const float* lf = logits + (long long)b * NTOK;
    if (tid == 0) scnt = 0;
    __syncthreads();
    int local = 0;
    for (int n = tid; n < NTOK; n += 256) {
        float v = lf[n]; vals[n] = v; if (v > 0.0f) local++;
    }
    atomicAdd(&scnt, local);
    __syncthreads();
    for (int it = 0; it < MIN_TOK; ++it) {
        float best = F32_NINF; int bi = 0;
        for (int n = tid; n < NTOK; n += 256) {
            float v = vals[n]; if (v > best) { best = v; bi = n; }
        }
        rv[tid] = best; ri[tid] = bi;
        __syncthreads();
        for (int s = 128; s > 0; s >>= 1) {
            if (tid < s && rv[tid + s] > rv[tid]) { rv[tid] = rv[tid + s]; ri[tid] = ri[tid + s]; }
            __syncthreads();
        }
        if (tid == 0) { kth_s = rv[0]; vals[ri[0]] = F32_NINF; }
        __syncthreads();
    }
    float kth = kth_s;
    int cnt = scnt;
    for (int n = tid; n < NTOK; n += 256) {
        float v = lf[n];
        valid[(long long)b * NTOK + n] = (cnt >= MIN_TOK) ? (v > 0.0f) : (v >= kth);
    }
}

// ---------------------------------------------------------------------------
// refined += x2
// ---------------------------------------------------------------------------
__global__ __launch_bounds__(256)
void add_kernel(float* __restrict__ dst, const float* __restrict__ src, long long n)
{
    long long i = (long long)blockIdx.x * 256 + threadIdx.x;
    if (i < n) dst[i] += src[i];
}

// ---------------------------------------------------------------------------
// final: out[b,c,n] = valid[b,n] ? refined[b,n,c] : flat_ir+flat_vis ; +scalar
// ---------------------------------------------------------------------------
__global__ __launch_bounds__(256)
void final_kernel(const float* __restrict__ refined, const float* __restrict__ flat_ir,
                  const float* __restrict__ flat_vis, const int* __restrict__ valid,
                  float* __restrict__ out)
{
    long long i = (long long)blockIdx.x * 256 + threadIdx.x;
    const long long TOT = (long long)BATCH * DIM * NTOK;
    if (i == 0) out[TOT] = 0.0f;   // second tuple element (scalar 0.0)
    if (i >= TOT) return;
    int n = (int)(i % NTOK);
    long long t = i / NTOK;
    int c = (int)(t % DIM);
    int b = (int)(t / DIM);
    long long row = (long long)b * NTOK + n;
    float v = valid[row] ? refined[row * DIM + c]
                         : (flat_ir[row * DIM + c] + flat_vis[row * DIM + c]);
    out[i] = v;
}

// ---------------------------------------------------------------------------
// Host side
// ---------------------------------------------------------------------------
struct MixP {
    const float *ln_g, *ln_b, *in_w, *in_b, *out_w, *out_b, *f1_w, *f1_b, *f2_w, *f2_b;
};

template<int EPI, bool KCONTIG>
static inline void launch_gemm(const float* A, int lda,
                               const float* B, long long ldbn, long long ldbk,
                               float* C, int ldc,
                               const float* bias, const float* res,
                               int M, int N, int K, float scale, hipStream_t stream)
{
    dim3 grid((N + 127) / 128, (M + 127) / 128);
    gemm_wmma<EPI, KCONTIG><<<grid, 256, 0, stream>>>(A, lda, B, ldbn, ldbk, C, ldc,
                                                      bias, res, M, N, K, scale);
}

static void run_mixer(const float* x, const MixP& p, float* outbuf,
                      float* T_XN, float* T_O, float* T_X1, float* S_ATT, float* T_U,
                      const int* VALID, hipStream_t stream)
{
    float* QKV = T_U;   // 8192 x 1152 (dead before FFN hidden reuses T_U)
    float* T_H = T_U;   // 8192 x 1536

    // LN1
    ln_kernel<<<ROWS, 128, 0, stream>>>(x, p.ln_g, p.ln_b, T_XN);
    // QKV projection
    launch_gemm<EPI_BIAS, true>(T_XN, DIM, p.in_w, DIM, 1, QKV, QKV_DIM,
                                p.in_b, nullptr, ROWS, QKV_DIM, DIM, 1.0f, stream);
    // attention, per batch / head (scores materialized per batch in L2)
    const float scl = 0.10206207261596577f;  // 1/sqrt(96)
    for (int b = 0; b < BATCH; ++b) {
        const float* qkv_b = QKV + (long long)b * NTOK * QKV_DIM;
        for (int h = 0; h < HEADS; ++h) {
            launch_gemm<EPI_MASK, true>(qkv_b + h * HDIM, QKV_DIM,           // Q
                                        qkv_b + DIM + h * HDIM, QKV_DIM, 1,  // K as W[n,k]
                                        S_ATT + (long long)h * NTOK * NTOK, NTOK,
                                        (const float*)(VALID + (long long)b * NTOK), nullptr,
                                        NTOK, NTOK, HDIM, scl, stream);
        }
        softmax_kernel<<<HEADS * NTOK, 256, 0, stream>>>(S_ATT);
        for (int h = 0; h < HEADS; ++h) {
            launch_gemm<EPI_PLAIN, false>(S_ATT + (long long)h * NTOK * NTOK, NTOK,  // P
                                          qkv_b + 2 * DIM + h * HDIM, 1, QKV_DIM,    // V^T
                                          T_O + (long long)b * NTOK * DIM + h * HDIM, DIM,
                                          nullptr, nullptr, NTOK, HDIM, NTOK, 1.0f, stream);
        }
    }
    // out projection + residual
    launch_gemm<EPI_BIAS, true>(T_O, DIM, p.out_w, DIM, 1, T_X1, DIM,
                                p.out_b, x, ROWS, DIM, DIM, 1.0f, stream);
    // LN2
    ln_kernel<<<ROWS, 128, 0, stream>>>(T_X1, p.ln_g, p.ln_b, T_XN);
    // FFN
    launch_gemm<EPI_GELU, true>(T_XN, DIM, p.f1_w, DIM, 1, T_H, FFN_HID,
                                p.f1_b, nullptr, ROWS, FFN_HID, DIM, 1.0f, stream);
    launch_gemm<EPI_BIAS, true>(T_H, FFN_HID, p.f2_w, FFN_HID, 1, outbuf, DIM,
                                p.f2_b, T_X1, ROWS, DIM, FFN_HID, 1.0f, stream);
}

extern "C" void kernel_launch(void* const* d_in, const int* in_sizes, int n_in,
                              void* d_out, int out_size, void* d_ws, size_t ws_size,
                              hipStream_t stream)
{
    (void)in_sizes; (void)n_in; (void)out_size; (void)ws_size;

    // setup_inputs() insertion order (dicts flattened in insertion order)
    const float* f_ir  = (const float*)d_in[0];
    const float* f_vis = (const float*)d_in[1];
    const float* a_w1  = (const float*)d_in[2];
    const float* a_b1  = (const float*)d_in[3];
    const float* a_w2  = (const float*)d_in[4];
    const float* a_b2  = (const float*)d_in[5];
    MixP pir  = { (const float*)d_in[6],  (const float*)d_in[7],  (const float*)d_in[8],
                  (const float*)d_in[9],  (const float*)d_in[10], (const float*)d_in[11],
                  (const float*)d_in[12], (const float*)d_in[13], (const float*)d_in[14],
                  (const float*)d_in[15] };
    MixP pvis = { (const float*)d_in[16], (const float*)d_in[17], (const float*)d_in[18],
                  (const float*)d_in[19], (const float*)d_in[20], (const float*)d_in[21],
                  (const float*)d_in[22], (const float*)d_in[23], (const float*)d_in[24],
                  (const float*)d_in[25] };

    float* ws = (float*)d_ws;
    const long long SF = (long long)ROWS * DIM;   // 3,145,728
    float* FLAT_IR  = ws;
    float* FLAT_VIS = ws + SF;
    float* REFINED  = ws + 2 * SF;
    float* LOGITS   = ws + 3 * SF;
    int*   VALID    = (int*)(ws + 3 * SF + ROWS);
    float* SCR      = ws + 3 * SF + 2 * ROWS;
    // agent scratch (dead before mixers) overlays mixer scratch
    float* XAGENT = SCR;                                      // 8192*768
    float* HAGENT = SCR + (long long)ROWS * AGENT_IN;         // 8192*512
    float* T_XN   = SCR;                                      // 8192*384
    float* T_O    = SCR + SF;                                 // 8192*384
    float* T_X1   = SCR + 2 * SF;                             // 8192*384
    float* T_X2   = SCR + 3 * SF;                             // 8192*384
    float* S_ATT  = SCR + 4 * SF;                             // 4*1024*1024
    float* T_U    = SCR + 4 * SF + (long long)HEADS * NTOK * NTOK;  // max(QKV, FFN_HID)

    const long long TOT = (long long)BATCH * DIM * NTOK;
    const int nb = (int)((TOT + 255) / 256);

    // 1) flatten + agent input
    prep_kernel<<<nb, 256, 0, stream>>>(f_ir, f_vis, FLAT_IR, FLAT_VIS, XAGENT);
    // 2) agent MLP hidden (SiLU) via WMMA
    launch_gemm<EPI_SILU, true>(XAGENT, AGENT_IN, a_w1, AGENT_IN, 1, HAGENT, AGENT_HID,
                                a_b1, nullptr, ROWS, AGENT_HID, AGENT_IN, 1.0f, stream);
    // 3) logits + 4) valid mask (top-k fallback)
    logits_kernel<<<ROWS / 8, 256, 0, stream>>>(HAGENT, a_w2, a_b2, LOGITS);
    valid_kernel<<<BATCH, 256, 0, stream>>>(LOGITS, VALID);
    // 5) mixers
    run_mixer(FLAT_IR,  pir,  REFINED, T_XN, T_O, T_X1, S_ATT, T_U, VALID, stream);
    run_mixer(FLAT_VIS, pvis, T_X2,    T_XN, T_O, T_X1, S_ATT, T_U, VALID, stream);
    add_kernel<<<nb, 256, 0, stream>>>(REFINED, T_X2, SF);
    // 6) masked combine back to (B,C,H,W) + trailing scalar
    final_kernel<<<nb, 256, 0, stream>>>(REFINED, FLAT_IR, FLAT_VIS, VALID, (float*)d_out);
}
